// GAT_62010737819882
// MI455X (gfx1250) — compile-verified, matrix operand order checked by reading, not run on hardware
//
#include <hip/hip_runtime.h>
#include <math.h>

// ---------------- problem constants (from reference) ----------------
#define NN      50000      // nodes
#define EE      800000     // raw edges
#define ET      850000     // edges + self loops
#define FIN     128
#define C1      256        // layer-1 output channels (4 heads x 64)
#define H1      4
#define FO1     64
#define C2      128        // layer-2 output channels (1 head x 128)
#define H2      1
#define FO2     128
#define NB      512        // graphs
#define NEG_SLOPE 0.2f

typedef __attribute__((ext_vector_type(16))) _Float16 v16h;
typedef __attribute__((ext_vector_type(8)))  _Float16 v8h;
typedef __attribute__((ext_vector_type(8)))  float    v8f;

// ---------------- utility ----------------
__global__ void fill_f32(float* __restrict__ p, float v, long n) {
    long i = (long)blockIdx.x * blockDim.x + threadIdx.x;
    if (i < n) p[i] = v;
}

__global__ void cast_f16(const float* __restrict__ in, _Float16* __restrict__ out, long n) {
    long i = (long)blockIdx.x * blockDim.x + threadIdx.x;
    if (i < n) out[i] = (_Float16)in[i];
}

// WT[fo, k] = (f16) W[k, fo]  -- transpose+convert so B fragments load contiguously
__global__ void transpose_cast_f16(const float* __restrict__ W, _Float16* __restrict__ WT,
                                   int K, int FO) {
    int i = blockIdx.x * blockDim.x + threadIdx.x;
    if (i >= K * FO) return;
    int k = i / FO, fo = i % FO;
    WT[(long)fo * K + k] = (_Float16)W[i];
}

__device__ __forceinline__ void atomicMaxF(float* addr, float val) {
    unsigned int* ia = (unsigned int*)addr;
    unsigned int old = *ia;
    while (true) {
        float f = __uint_as_float(old);
        if (f >= val) return;
        unsigned int assumed = old;
        old = atomicCAS(ia, assumed, __float_as_uint(val));
        if (old == assumed) return;
    }
}

__device__ __forceinline__ int edge_src(const int* ei, int e) {
    return (e < EE) ? ei[e] : (e - EE);
}
__device__ __forceinline__ int edge_dst(const int* ei, int e) {
    return (e < EE) ? ei[EE + e] : (e - EE);
}

// ---------------- WMMA GEMM:  C[N,FO] = A[N,K] * W[K,FO]
// A is pre-converted f16 row-major [N,K]; WT is f16 transposed [FO,K].
// One wave32 per (16-row x CPW*16-col) strip; A fragment reused CPW times.
template<int K, int FO, int CPW>
__global__ void gemm_wmma(const _Float16* __restrict__ A, const _Float16* __restrict__ WT,
                          float* __restrict__ C, int nrows) {
    const int lane = threadIdx.x & 31;
    const int wave = threadIdx.x >> 5;
    const int colGroups = FO / (16 * CPW);
    const int tile = blockIdx.x * (blockDim.x >> 5) + wave;   // wave-uniform
    const int numRowTiles = nrows / 16;
    if (tile >= numRowTiles * colGroups) return;              // wave-uniform exit
    const int rowTile = tile / colGroups;
    const int colG    = tile % colGroups;

    const int m    = lane & 15;          // A row within tile / B,C column within tile
    const int half = lane >> 4;          // lane group selects K sub-block
    const long row = (long)rowTile * 16 + m;
    const _Float16* ap = A + row * K + half * 8;

    v8f acc[CPW];
#pragma unroll
    for (int c = 0; c < CPW; ++c) acc[c] = (v8f){};

    for (int kb = 0; kb < K; kb += 32) {
        // A fragment: K = {k0..k0+7, k0+16..k0+23}, k0 = kb + half*8
        v8h alo = *(const v8h*)(ap + kb);
        v8h ahi = *(const v8h*)(ap + kb + 16);
        v16h a;
#pragma unroll
        for (int i = 0; i < 8; ++i) { a[i] = alo[i]; a[i + 8] = ahi[i]; }

#pragma unroll
        for (int c = 0; c < CPW; ++c) {
            const int col = (colG * CPW + c) * 16 + m;
            const _Float16* wp = WT + (long)col * K + half * 8 + kb;
            v8h blo = *(const v8h*)wp;
            v8h bhi = *(const v8h*)(wp + 16);
            v16h b;
#pragma unroll
            for (int i = 0; i < 8; ++i) { b[i] = blo[i]; b[i + 8] = bhi[i]; }
            acc[c] = __builtin_amdgcn_wmma_f32_16x16x32_f16(
                /*neg_a=*/false, a, /*neg_b=*/false, b,
                /*c_mod=*/(short)0, acc[c], /*reuse_a=*/false, /*reuse_b=*/false);
        }
    }

    // C/D layout: lane holds column m; VGPR r holds row (r + half*8) -> coalesced row stores
#pragma unroll
    for (int c = 0; c < CPW; ++c) {
        float* cp = C + ((long)rowTile * 16) * FO + (colG * CPW + c) * 16 + m;
#pragma unroll
        for (int r = 0; r < 8; ++r)
            cp[(long)(r + half * 8) * FO] = acc[c][r];
    }
}

// ---------------- attention logits: alpha_s[n,h] = <h[n,h,:], a_src[h,:]> ----------------
template<int H, int FOH>
__global__ void alpha_kernel(const float* __restrict__ h,
                             const float* __restrict__ a_src,
                             const float* __restrict__ a_dst,
                             float* __restrict__ as, float* __restrict__ ad) {
    int tid = blockIdx.x * blockDim.x + threadIdx.x;
    if (tid >= NN * H) return;
    int n = tid / H, hh = tid % H;
    const float* hp = h + (long)n * (H * FOH) + hh * FOH;
    const float* sp = a_src + hh * FOH;
    const float* dp = a_dst + hh * FOH;
    float s = 0.f, d = 0.f;
    for (int f = 0; f < FOH; ++f) { s += hp[f] * sp[f]; d += hp[f] * dp[f]; }
    as[tid] = s; ad[tid] = d;
}

// ---------------- edge pass 1: segment max of leaky-relu logits ----------------
template<int H>
__global__ void edge_max(const int* __restrict__ ei,
                         const float* __restrict__ as, const float* __restrict__ ad,
                         float* __restrict__ m) {
    int tid = blockIdx.x * blockDim.x + threadIdx.x;
    if (tid >= ET * H) return;
    int e = tid / H, hh = tid % H;
    int s = edge_src(ei, e), d = edge_dst(ei, e);
    float v = as[s * H + hh] + ad[d * H + hh];
    v = v > 0.f ? v : NEG_SLOPE * v;
    atomicMaxF(&m[d * H + hh], v);
}

// ---------------- edge pass 2: exp(e - m[dst]) and segment sum ----------------
template<int H>
__global__ void edge_exp(const int* __restrict__ ei,
                         const float* __restrict__ as, const float* __restrict__ ad,
                         const float* __restrict__ m,
                         float* __restrict__ exbuf, float* __restrict__ den) {
    int tid = blockIdx.x * blockDim.x + threadIdx.x;
    if (tid >= ET * H) return;
    int e = tid / H, hh = tid % H;
    int s = edge_src(ei, e), d = edge_dst(ei, e);
    float v = as[s * H + hh] + ad[d * H + hh];
    v = v > 0.f ? v : NEG_SLOPE * v;
    float ex = __expf(v - m[d * H + hh]);
    exbuf[tid] = ex;
    atomicAdd(&den[d * H + hh], ex);
}

// ---------------- edge pass 3: out[dst] += h[src] * alpha   (one wave per edge) ----------------
template<int H, int C>
__global__ void edge_agg(const int* __restrict__ ei,
                         const float* __restrict__ h,
                         const float* __restrict__ exbuf, const float* __restrict__ den,
                         float* __restrict__ out) {
    const int lane = threadIdx.x & 31;
    const int wave = threadIdx.x >> 5;
    const int e = blockIdx.x * (blockDim.x >> 5) + wave;
    if (e >= ET) return;
    const int s = edge_src(ei, e), d = edge_dst(ei, e);
    const int FPL = C / 32;                 // features per lane (8 or 4)
    const int f0 = lane * FPL;
    const int FOH = C / H;
    const int hh = f0 / FOH;                // head is constant over this lane's features
    const float alpha = exbuf[(long)e * H + hh] / den[d * H + hh];
    const float* hp = h + (long)s * C + f0;
    float* op = out + (long)d * C + f0;
#pragma unroll
    for (int j = 0; j < FPL; ++j)
        atomicAdd(&op[j], hp[j] * alpha);
}

// ---------------- bias + ELU -> f16 (feeds next WMMA GEMM) ----------------
template<int C>
__global__ void elu_bias_f16(const float* __restrict__ x, const float* __restrict__ b,
                             _Float16* __restrict__ out) {
    long i = (long)blockIdx.x * blockDim.x + threadIdx.x;
    if (i >= (long)NN * C) return;
    float v = x[i] + b[i % C];
    out[i] = (_Float16)(v > 0.f ? v : (__expf(v) - 1.f));
}

// ---------------- bias + ELU in place (f32, final layer) ----------------
template<int C>
__global__ void elu_bias_f32(float* __restrict__ x, const float* __restrict__ b) {
    long i = (long)blockIdx.x * blockDim.x + threadIdx.x;
    if (i >= (long)NN * C) return;
    float v = x[i] + b[i % C];
    x[i] = v > 0.f ? v : (__expf(v) - 1.f);
}

// ---------------- global mean pool ----------------
__global__ void pool_add(const float* __restrict__ h, const int* __restrict__ batch,
                         float* __restrict__ g) {
    long i = (long)blockIdx.x * blockDim.x + threadIdx.x;
    if (i >= (long)NN * C2) return;
    int n = (int)(i / C2), c = (int)(i % C2);
    atomicAdd(&g[(long)batch[n] * C2 + c], h[i]);
}
__global__ void pool_cnt(const int* __restrict__ batch, float* __restrict__ cnt) {
    int n = blockIdx.x * blockDim.x + threadIdx.x;
    if (n >= NN) return;
    atomicAdd(&cnt[batch[n]], 1.0f);
}

// ---------------- readout: sigmoid(mean(g) @ Wl + bl) ----------------
__global__ void readout(const float* __restrict__ g, const float* __restrict__ cnt,
                        const float* __restrict__ Wl, const float* __restrict__ bl,
                        float* __restrict__ out) {
    int b = blockIdx.x * blockDim.x + threadIdx.x;
    if (b >= NB) return;
    float inv = 1.0f / fmaxf(cnt[b], 1.0f);
    float z = bl[0];
    const float* gp = g + (long)b * C2;
    for (int c = 0; c < C2; ++c) z += gp[c] * inv * Wl[c];
    out[b] = 1.0f / (1.0f + __expf(-z));
}

// ---------------- host launcher ----------------
static inline int cdiv(long a, int b) { return (int)((a + b - 1) / b); }

extern "C" void kernel_launch(void* const* d_in, const int* in_sizes, int n_in,
                              void* d_out, int out_size, void* d_ws, size_t ws_size,
                              hipStream_t stream) {
    const float* x      = (const float*)d_in[0];
    const int*   ei     = (const int*)  d_in[1];
    const int*   batch  = (const int*)  d_in[2];
    const float* W1     = (const float*)d_in[3];
    const float* a_src1 = (const float*)d_in[4];
    const float* a_dst1 = (const float*)d_in[5];
    const float* b1     = (const float*)d_in[6];
    const float* W2     = (const float*)d_in[7];
    const float* a_src2 = (const float*)d_in[8];
    const float* a_dst2 = (const float*)d_in[9];
    const float* b2     = (const float*)d_in[10];
    const float* Wl     = (const float*)d_in[11];
    const float* bl     = (const float*)d_in[12];
    float* out = (float*)d_out;

    // workspace layout (floats; all region sizes are multiples of 4 floats -> 16B aligned)
    float* ws   = (float*)d_ws;
    float* h1   = ws;                       // N*C1 f32 (GEMM1 out; GEMM2 out reuses first N*C2)
    float* agg1 = h1   + (long)NN * C1;     // N*C1 f32 (layer-1 aggregation; later reused as out2)
    float* as1  = agg1 + (long)NN * C1;     // N*H1
    float* ad1  = as1  + (long)NN * H1;
    float* m1   = ad1  + (long)NN * H1;
    float* den1 = m1   + (long)NN * H1;
    float* as2  = den1 + (long)NN * H1;     // N (H2==1)
    float* ad2  = as2  + NN;
    float* m2   = ad2  + NN;
    float* den2 = m2   + NN;
    float* exb  = den2 + NN;                // ET*H1 (reused for layer 2, needs ET*H2)
    float* g    = exb  + (long)ET * H1;     // B*C2
    float* cnt  = g    + (long)NB * C2;     // B
    _Float16* a16  = (_Float16*)(cnt + NB);         // N*C1 halves (x16 then act1_16)
    _Float16* wt16 = a16 + (long)NN * C1;           // up to 32768 halves (W^T f16)
    float* h2   = h1;                       // reuse
    float* out2 = agg1;                     // reuse (agg1 consumed by elu_bias_f16 first)

    const int T = 256;

    // ================= layer 1 =================
    cast_f16<<<cdiv((long)NN * FIN, T), T, 0, stream>>>(x, a16, (long)NN * FIN);
    transpose_cast_f16<<<cdiv(FIN * C1, T), T, 0, stream>>>(W1, wt16, FIN, C1);
    gemm_wmma<FIN, C1, 4><<<cdiv((long)(NN / 16) * (C1 / 64), 8), T, 0, stream>>>(a16, wt16, h1, NN);

    alpha_kernel<H1, FO1><<<cdiv((long)NN * H1, T), T, 0, stream>>>(h1, a_src1, a_dst1, as1, ad1);

    fill_f32<<<cdiv((long)NN * H1, T), T, 0, stream>>>(m1, -1e30f, (long)NN * H1);
    fill_f32<<<cdiv((long)NN * H1, T), T, 0, stream>>>(den1, 0.f, (long)NN * H1);
    fill_f32<<<cdiv((long)NN * C1, T), T, 0, stream>>>(agg1, 0.f, (long)NN * C1);

    edge_max<H1><<<cdiv((long)ET * H1, T), T, 0, stream>>>(ei, as1, ad1, m1);
    edge_exp<H1><<<cdiv((long)ET * H1, T), T, 0, stream>>>(ei, as1, ad1, m1, exb, den1);
    edge_agg<H1, C1><<<cdiv(ET, 8), T, 0, stream>>>(ei, h1, exb, den1, agg1);
    elu_bias_f16<C1><<<cdiv((long)NN * C1, T), T, 0, stream>>>(agg1, b1, a16);

    // ================= layer 2 =================
    transpose_cast_f16<<<cdiv(C1 * C2, T), T, 0, stream>>>(W2, wt16, C1, C2);
    gemm_wmma<C1, C2, 4><<<cdiv((long)(NN / 16) * (C2 / 64), 8), T, 0, stream>>>(a16, wt16, h2, NN);

    alpha_kernel<H2, FO2><<<cdiv((long)NN * H2, T), T, 0, stream>>>(h2, a_src2, a_dst2, as2, ad2);

    fill_f32<<<cdiv((long)NN * H2, T), T, 0, stream>>>(m2, -1e30f, (long)NN * H2);
    fill_f32<<<cdiv((long)NN * H2, T), T, 0, stream>>>(den2, 0.f, (long)NN * H2);
    fill_f32<<<cdiv((long)NN * C2, T), T, 0, stream>>>(out2, 0.f, (long)NN * C2);

    edge_max<H2><<<cdiv((long)ET * H2, T), T, 0, stream>>>(ei, as2, ad2, m2);
    edge_exp<H2><<<cdiv((long)ET * H2, T), T, 0, stream>>>(ei, as2, ad2, m2, exb, den2);
    edge_agg<H2, C2><<<cdiv(ET, 8), T, 0, stream>>>(ei, h2, exb, den2, out2);
    elu_bias_f32<C2><<<cdiv((long)NN * C2, T), T, 0, stream>>>(out2, b2);

    // ================= pooling + readout =================
    fill_f32<<<cdiv((long)NB * C2, T), T, 0, stream>>>(g, 0.f, (long)NB * C2);
    fill_f32<<<cdiv(NB, T), T, 0, stream>>>(cnt, 0.f, NB);
    pool_add<<<cdiv((long)NN * C2, T), T, 0, stream>>>(out2, batch, g);
    pool_cnt<<<cdiv(NN, T), T, 0, stream>>>(batch, cnt);
    readout<<<cdiv(NB, T), T, 0, stream>>>(g, cnt, Wl, bl, out);
}
